// Attention_34325378629934
// MI455X (gfx1250) — matrix operand order repaired and checked by reading, not compiled, not run on hardware
//
#include <hip/hip_runtime.h>
#include <hip/hip_bf16.h>

// ---------------------------------------------------------------------------
// MDTA (transposed attention) for MI455X / gfx1250, wave32 + WMMA bf16.
// All GEMM-like stages use v_wmma_f32_16x16x32_bf16 (f32 accumulate).
// L2 norms of q/k are folded into the 24x24 score matrix so normalized q/k
// are never materialized. Intermediates are bf16 to halve HBM traffic and
// are kept L2-resident; strictly-streaming tensors use non-temporal hints.
// ---------------------------------------------------------------------------

typedef __attribute__((ext_vector_type(16))) __bf16        v16bf;
typedef __attribute__((ext_vector_type(8)))  float         v8f;
typedef __attribute__((ext_vector_type(8)))  unsigned int  v8u;

#define B_      4
#define C_      192
#define C3_     576
#define NH_     8
#define HC_     24
#define HW_     65536
#define KSPLIT_ 8
#define EPS_    1e-12f

__device__ __forceinline__ unsigned short f2bf(float f) {
  unsigned int u = __float_as_uint(f);
  u += 0x7fffu + ((u >> 16) & 1u);            // round-to-nearest-even
  return (unsigned short)(u >> 16);
}
__device__ __forceinline__ float bf2f(unsigned short h) {
  return __uint_as_float(((unsigned int)h) << 16);
}
// Build a 16 x bf16 WMMA operand register block from two 16B chunks.
__device__ __forceinline__ v16bf pack16(uint4 lo, uint4 hi) {
  v8u u;
  u[0]=lo.x; u[1]=lo.y; u[2]=lo.z; u[3]=lo.w;
  u[4]=hi.x; u[5]=hi.y; u[6]=hi.z; u[7]=hi.w;
  return __builtin_bit_cast(v16bf, u);
}

// ---------------------------------------------------------------------------
__global__ void cvt_kernel(const float* __restrict__ s,
                           unsigned short* __restrict__ d, int n) {
  for (int i = blockIdx.x * blockDim.x + threadIdx.x; i < n;
       i += gridDim.x * blockDim.x)
    d[i] = f2bf(__builtin_nontemporal_load(s + i));   // one-shot read
}

__global__ void zero_kernel(float* __restrict__ p, int n) {
  int i = blockIdx.x * blockDim.x + threadIdx.x;
  if (i < n) p[i] = 0.f;
}

// ---------------------------------------------------------------------------
// 1x1 conv as GEMM:  out[n][co][px] = sum_ci W[co][ci] * X[n][ci][px] + b[co]
// K = 192 (6 wmma k-steps). Block = 256 thr (8 waves).
// Block tile: M = 64 co (4 mtiles looped per wave) x N = 128 px (1 col/wave).
// X tile is transposed through LDS so the B operand reads contiguous K;
// each staged tile feeds 4 WMMAs per wave. STORE_F32 selected at compile
// time so the epilogue is a straight store clause (no per-element branches).
// grid: (HW/128, Cout/64, B)
// ---------------------------------------------------------------------------
template <bool STORE_F32>
__global__ void __launch_bounds__(256)
pw_gemm_kernel(const unsigned short* __restrict__ Xbf,  // [B][192][HW]
               const unsigned short* __restrict__ Wbf,  // [Cout][192]
               const float* __restrict__ bias,          // [Cout]
               unsigned short* __restrict__ outBf,      // if !STORE_F32
               float* __restrict__ outF32,              // if STORE_F32
               int Cout)
{
  __shared__ __align__(16) unsigned short xT[128][40];  // [px][ci], 80B rows
  const int tid    = threadIdx.x;
  const int lane   = tid & 31;
  const int wave   = tid >> 5;
  const int laneN  = lane & 15;
  const int laneHi = lane >> 4;
  const int pxbase = blockIdx.x * 128;
  const int mbase  = blockIdx.y * 64;
  const int n      = blockIdx.z;

  const unsigned short* xb = Xbf + (size_t)n * C_ * HW_;
  v8f acc[4] = {};

  for (int k0 = 0; k0 < C_; k0 += 32) {
    __syncthreads();
    {
      // stage 32(ci) x 128(px) tile, transposed: coalesced 32B global reads
      int ci = tid >> 3;
      int ps = (tid & 7) * 16;
      const unsigned short* src = xb + (size_t)(k0 + ci) * HW_ + pxbase + ps;
      __builtin_prefetch(src + 32 * HW_, 0, 1);
      uint4 a = *(const uint4*)(src);
      uint4 b = *(const uint4*)(src + 8);
      unsigned int ww[8] = {a.x, a.y, a.z, a.w, b.x, b.y, b.z, b.w};
#pragma unroll
      for (int i = 0; i < 8; ++i) {
        xT[ps + 2*i    ][ci] = (unsigned short)(ww[i] & 0xffffu);
        xT[ps + 2*i + 1][ci] = (unsigned short)(ww[i] >> 16);
      }
    }
    __syncthreads();

    // B operand: transposed X rows (contiguous K per pixel)
    const unsigned short* xrow = &xT[wave * 16 + laneN][laneHi * 16];
    v16bf Bm = pack16(*(const uint4*)(xrow), *(const uint4*)(xrow + 8));

#pragma unroll
    for (int mi = 0; mi < 4; ++mi) {
      const unsigned short* wrow =
          Wbf + (size_t)(mbase + mi * 16 + laneN) * C_ + k0 + laneHi * 16;
      v16bf A = pack16(*(const uint4*)(wrow), *(const uint4*)(wrow + 8));
      acc[mi] = __builtin_amdgcn_wmma_f32_16x16x32_bf16(
          false, A, false, Bm, (short)0, acc[mi], false, false);
    }
  }

  const int px = pxbase + wave * 16 + laneN;
#pragma unroll
  for (int mi = 0; mi < 4; ++mi) {
#pragma unroll
    for (int r = 0; r < 8; ++r) {
      int co = mbase + mi * 16 + laneHi * 8 + r;
      float v = acc[mi][r] + bias[co];
      size_t off = ((size_t)n * Cout + co) * HW_ + px;
      if (STORE_F32) __builtin_nontemporal_store(v, outF32 + off); // final out
      else           outBf[off] = f2bf(v);                         // L2-hot
    }
  }
}

// ---------------------------------------------------------------------------
// Depthwise 3x3 (SAME) + bias; also accumulates sum(q^2), sum(k^2) per
// (n, channel) for the folded L2 normalization.
// grid: (B*576, HW/2048), 256 threads.
// ---------------------------------------------------------------------------
__global__ void __launch_bounds__(256)
dwconv_kernel(const unsigned short* __restrict__ qkvBf,
              const float* __restrict__ dww,   // [576][3][3]
              const float* __restrict__ dwb,   // [576]
              unsigned short* __restrict__ outBf,
              float* __restrict__ norms)       // [2][B][192] sumsq
{
  const int bc    = blockIdx.x;
  const int n     = bc / C3_;
  const int ch    = bc % C3_;
  const int pbase = blockIdx.y * 2048;
  const int tid   = threadIdx.x;

  float w[9];
#pragma unroll
  for (int i = 0; i < 9; ++i) w[i] = dww[ch * 9 + i];
  const float bias = dwb[ch];

  const unsigned short* src = qkvBf + ((size_t)n * C3_ + ch) * HW_;
  unsigned short*       dst = outBf + ((size_t)n * C3_ + ch) * HW_;

  float ss = 0.f;
  for (int i = 0; i < 8; ++i) {
    int p = pbase + i * 256 + tid;
    int y = p >> 8, x = p & 255;
    float a = bias;
#pragma unroll
    for (int dy = -1; dy <= 1; ++dy) {
      int yy = y + dy;
      if (yy < 0 || yy > 255) continue;
#pragma unroll
      for (int dx = -1; dx <= 1; ++dx) {
        int xx = x + dx;
        if (xx < 0 || xx > 255) continue;
        a += w[(dy + 1) * 3 + (dx + 1)] * bf2f(src[yy * 256 + xx]);
      }
    }
    dst[p] = f2bf(a);
    if (ch < 2 * C_) ss += a * a;
  }

  if (ch < 2 * C_) {
    __shared__ float red[256];
    red[tid] = ss;
    __syncthreads();
    for (int s = 128; s > 0; s >>= 1) {
      if (tid < s) red[tid] += red[tid + s];
      __syncthreads();
    }
    if (tid == 0) {
      int qi = (ch < C_) ? 0 : 1;
      int c  = ch - qi * C_;
      atomicAdd(&norms[qi * (B_ * C_) + n * C_ + c], red[0]);
    }
  }
}

// ---------------------------------------------------------------------------
// Partial scores: S[c][d] += sum_{s in K-slice} q[c][s]*k[d][s]
// (24x24 padded to 32x32). grid: (B*8 heads, KSPLIT). Each block: 8 waves
// split its K-slice; 2x2 WMMA tiles; cross-wave LDS reduce; atomicAdd into
// a per-(b,head) 32x32 f32 buffer. 256 blocks keep the WGPs fed on what is
// a pure HBM-streaming reduction (201 MB of q+k).
// ---------------------------------------------------------------------------
__global__ void __launch_bounds__(256)
attn_scores_partial_kernel(const unsigned short* __restrict__ dwBf, // [B][576][HW]
                           float* __restrict__ Sbuf)                // [B*8][32][32]
{
  const int bh    = blockIdx.x;
  const int n     = bh / NH_, head = bh % NH_;
  const int kseg  = blockIdx.y * (HW_ / KSPLIT_);
  const int tid   = threadIdx.x;
  const int lane  = tid & 31, wave = tid >> 5;
  const int laneN = lane & 15, laneHi = lane >> 4;

  const unsigned short* qb = dwBf + ((size_t)n * C3_ +       head * HC_) * HW_;
  const unsigned short* kb = dwBf + ((size_t)n * C3_ + C_ +  head * HC_) * HW_;

  v8f acc[2][2] = {};
  const uint4 z4 = make_uint4(0u, 0u, 0u, 0u);

  for (int k0 = kseg + wave * 32; k0 < kseg + HW_ / KSPLIT_; k0 += 256) {
    v16bf A[2], Bt[2];
#pragma unroll
    for (int ti = 0; ti < 2; ++ti) {
      int c = ti * 16 + laneN;
      uint4 a0 = z4, a1 = z4;
      if (c < HC_) {
        const unsigned short* p = qb + (size_t)c * HW_ + k0 + laneHi * 16;
        __builtin_prefetch(p + 256, 0, 1);
        a0 = *(const uint4*)(p);
        a1 = *(const uint4*)(p + 8);
      }
      A[ti] = pack16(a0, a1);
    }
#pragma unroll
    for (int tj = 0; tj < 2; ++tj) {
      int d = tj * 16 + laneN;
      uint4 b0 = z4, b1 = z4;
      if (d < HC_) {
        const unsigned short* p = kb + (size_t)d * HW_ + k0 + laneHi * 16;
        __builtin_prefetch(p + 256, 0, 1);
        b0 = *(const uint4*)(p);
        b1 = *(const uint4*)(p + 8);
      }
      Bt[tj] = pack16(b0, b1);
    }
#pragma unroll
    for (int ti = 0; ti < 2; ++ti)
#pragma unroll
      for (int tj = 0; tj < 2; ++tj)
        acc[ti][tj] = __builtin_amdgcn_wmma_f32_16x16x32_bf16(
            false, A[ti], false, Bt[tj], (short)0, acc[ti][tj], false, false);
  }

  // cross-wave reduction via LDS, then one atomicAdd per element
  __shared__ float red[8][1024];
#pragma unroll
  for (int ti = 0; ti < 2; ++ti)
#pragma unroll
    for (int tj = 0; tj < 2; ++tj)
#pragma unroll
      for (int r = 0; r < 8; ++r)
        red[wave][(ti * 2 + tj) * 256 + r * 32 + lane] = acc[ti][tj][r];
  __syncthreads();

  for (int e = tid; e < 1024; e += 256) {
    float s = 0.f;
#pragma unroll
    for (int w8 = 0; w8 < 8; ++w8) s += red[w8][e];
    int tile = e >> 8, slot = e & 255;
    int r = slot >> 5, ln = slot & 31;
    int ti = tile >> 1, tj = tile & 1;
    int m  = ti * 16 + (ln >> 4) * 8 + r;
    int nn = tj * 16 + (ln & 15);
    atomicAdd(&Sbuf[(size_t)bh * 1024 + m * 32 + nn], s);
  }
}

// ---------------------------------------------------------------------------
// Folded L2 normalization + scale + softmax over d (24). 32 blocks x 32 thr.
// ---------------------------------------------------------------------------
__global__ void __launch_bounds__(32)
attn_softmax_kernel(const float* __restrict__ Sbuf,   // [B*8][32][32]
                    const float* __restrict__ norms,  // [2][B][192] sumsq
                    const float* __restrict__ scale,  // [8]
                    float* __restrict__ attn)         // [B*8][24][24]
{
  const int bh = blockIdx.x;
  const int n  = bh / NH_, head = bh % NH_;
  const int c  = threadIdx.x;
  if (c >= HC_) return;

  float nq = fmaxf(sqrtf(norms[n * C_ + head * HC_ + c]), EPS_);
  float sc = scale[head];
  float row[HC_];
  float mx = -1e30f;
  for (int d = 0; d < HC_; ++d) {
    float nk = fmaxf(sqrtf(norms[B_ * C_ + n * C_ + head * HC_ + d]), EPS_);
    float v = Sbuf[(size_t)bh * 1024 + c * 32 + d] / (nq * nk) * sc;
    row[d] = v;
    mx = fmaxf(mx, v);
  }
  float sum = 0.f;
  for (int d = 0; d < HC_; ++d) { row[d] = __expf(row[d] - mx); sum += row[d]; }
  float inv = 1.f / sum;
  float* arow = attn + ((size_t)bh * HC_ + c) * HC_;
  for (int d = 0; d < HC_; ++d) arow[d] = row[d] * inv;
}

// ---------------------------------------------------------------------------
// out[c][px] = sum_d attn[c][d] * v[d][px]  (single K-step WMMA, K 24->32)
// grid: (HW/128, B*8), 256 threads (8 waves x 16 px each).
// ---------------------------------------------------------------------------
__global__ void __launch_bounds__(256)
attn_apply_kernel(const float* __restrict__ attn,            // [B*8][24][24]
                  const unsigned short* __restrict__ dwBf,   // [B][576][HW]
                  unsigned short* __restrict__ obuf)         // [B][192][HW]
{
  const int bh     = blockIdx.y;
  const int n      = bh / NH_, head = bh % NH_;
  const int pxbase = blockIdx.x * 128;
  const int tid    = threadIdx.x;
  const int lane   = tid & 31, wave = tid >> 5;
  const int laneN  = lane & 15, laneHi = lane >> 4;

  __shared__ __align__(16) unsigned short Ap[32][32];   // attn, bf16 padded
  __shared__ __align__(16) unsigned short vT[128][40];  // [px][d] transposed

  for (int e = tid; e < 1024; e += 256) {
    int c = e >> 5, d = e & 31;
    float v = (c < HC_ && d < HC_) ? attn[((size_t)bh * HC_ + c) * HC_ + d] : 0.f;
    Ap[c][d] = f2bf(v);
  }
  const unsigned short* vb =
      dwBf + ((size_t)n * C3_ + 2 * C_ + head * HC_) * HW_;
  for (int e = tid; e < 128 * 32; e += 256) {
    int d = e >> 7, px = e & 127;                 // px fastest -> coalesced
    vT[px][d] = (d < HC_) ? vb[(size_t)d * HW_ + pxbase + px]
                          : (unsigned short)0;
  }
  __syncthreads();

  const unsigned short* brow = &vT[wave * 16 + laneN][laneHi * 16];
  v16bf Bm = pack16(*(const uint4*)(brow), *(const uint4*)(brow + 8));

#pragma unroll
  for (int ti = 0; ti < 2; ++ti) {
    const unsigned short* arow = &Ap[ti * 16 + laneN][laneHi * 16];
    v16bf A = pack16(*(const uint4*)(arow), *(const uint4*)(arow + 8));
    v8f acc = {};
    acc = __builtin_amdgcn_wmma_f32_16x16x32_bf16(false, A, false, Bm,
                                                  (short)0, acc, false, false);
    int px = pxbase + wave * 16 + laneN;
#pragma unroll
    for (int r = 0; r < 8; ++r) {
      int c = ti * 16 + laneHi * 8 + r;
      if (c < HC_)
        obuf[((size_t)n * C_ + head * HC_ + c) * HW_ + px] = f2bf(acc[r]);
    }
  }
}

// ---------------------------------------------------------------------------
extern "C" void kernel_launch(void* const* d_in, const int* in_sizes, int n_in,
                              void* d_out, int out_size, void* d_ws,
                              size_t ws_size, hipStream_t stream) {
  const float* x      = (const float*)d_in[0];
  const float* qkv_w  = (const float*)d_in[1];
  const float* qkv_b  = (const float*)d_in[2];
  const float* dw_w   = (const float*)d_in[3];
  const float* dw_b   = (const float*)d_in[4];
  const float* scale  = (const float*)d_in[5];
  const float* proj_w = (const float*)d_in[6];
  const float* proj_b = (const float*)d_in[7];
  float* out = (float*)d_out;

  // workspace layout (all 256B aligned)
  char* ws = (char*)d_ws;
  size_t o = 0;
  auto take = [&](size_t bytes) {
    size_t r = o;
    o += (bytes + 255) & ~(size_t)255;
    return r;
  };
  unsigned short* xbf   = (unsigned short*)(ws + take((size_t)B_*C_*HW_*2));
  unsigned short* wqkv  = (unsigned short*)(ws + take((size_t)C3_*C_*2));
  unsigned short* wproj = (unsigned short*)(ws + take((size_t)C_*C_*2));
  unsigned short* qkvbf = (unsigned short*)(ws + take((size_t)B_*C3_*HW_*2));
  unsigned short* dwbf  = (unsigned short*)(ws + take((size_t)B_*C3_*HW_*2));
  float*          norms = (float*)(ws + take((size_t)2*B_*C_*4));
  float*          Sbuf  = (float*)(ws + take((size_t)B_*NH_*1024*4));
  float*          attnb = (float*)(ws + take((size_t)B_*NH_*HC_*HC_*4));
  unsigned short* obuf  = (unsigned short*)(ws + take((size_t)B_*C_*HW_*2));
  (void)ws_size; (void)in_sizes; (void)n_in; (void)out_size;

  // 1) precision conversion to bf16; zero accumulators
  cvt_kernel<<<4096, 256, 0, stream>>>(x, xbf, B_ * C_ * HW_);
  cvt_kernel<<<128, 256, 0, stream>>>(qkv_w, wqkv, C3_ * C_);
  cvt_kernel<<<64, 256, 0, stream>>>(proj_w, wproj, C_ * C_);
  zero_kernel<<<6, 256, 0, stream>>>(norms, 2 * B_ * C_);
  zero_kernel<<<128, 256, 0, stream>>>(Sbuf, B_ * NH_ * 1024);

  // 2) qkv 1x1 conv (WMMA GEMM, K=192)
  pw_gemm_kernel<false><<<dim3(HW_ / 128, C3_ / 64, B_), 256, 0, stream>>>(
      xbf, wqkv, qkv_b, qkvbf, nullptr, C3_);

  // 3) depthwise 3x3 + sumsq(q), sumsq(k)
  dwconv_kernel<<<dim3(B_ * C3_, HW_ / 2048), 256, 0, stream>>>(
      qkvbf, dw_w, dw_b, dwbf, norms);

  // 4) channel-attention scores, K split 8-ways (WMMA, K=65536 total)
  attn_scores_partial_kernel<<<dim3(B_ * NH_, KSPLIT_), 256, 0, stream>>>(
      dwbf, Sbuf);

  // 5) folded L2 norm + scale + softmax
  attn_softmax_kernel<<<B_ * NH_, 32, 0, stream>>>(Sbuf, norms, scale, attnb);

  // 6) out = attn @ v (WMMA, K=24 padded to 32)
  attn_apply_kernel<<<dim3(HW_ / 128, B_ * NH_), 256, 0, stream>>>(
      attnb, dwbf, obuf);

  // 7) proj 1x1 conv (WMMA GEMM, K=192) -> f32 output (non-temporal stores)
  pw_gemm_kernel<true><<<dim3(HW_ / 128, C_ / 64, B_), 256, 0, stream>>>(
      obuf, wproj, proj_b, nullptr, out, C_);
}
